// FoveatedQwen25VL_35519379538158
// MI455X (gfx1250) — matrix operand order repaired and checked by reading, not compiled, not run on hardware
//
#include <hip/hip_runtime.h>
#include <hip/hip_bf16.h>

// Dims: L=8, B=1, S=4096, D=2048, H=16, KVH=2, HD=128, Q=32
// Workspace layout (≈241 MB total):
//   cs     : 4096*128 float2      (4.0 MB)  cos/sin MROPE table
//   xbf    : 8*4096*2048 bf16     (134 MB)  RMSNorm'd activations
//   kpack  : 8*64*16*256 dword    (8.0 MB)  k_w in WMMA B-fragment layout
//   qpack  : 8*64*128*256 dword   (64  MB)  q_w in WMMA B-fragment layout
//   Kr     : 8*2*4096*128 bf16    (16  MB)  rope'd K cache
//   Qr     : 8*16*32*128 bf16     (1.0 MB)  rope'd Q cache
//   pq     : 4*8*32*2048 f32      (8.4 MB)  Q-proj K-split partials

typedef __attribute__((ext_vector_type(16))) __bf16 bf16x16;
typedef __attribute__((ext_vector_type(8)))  float  f32x8;

union Frag {
  uint4   u[2];
  bf16x16 v;
};

__device__ inline unsigned short f2bf(float f) {
  unsigned u = __float_as_uint(f);
  u += 0x7fffu + ((u >> 16) & 1u);   // round-to-nearest-even
  return (unsigned short)(u >> 16);
}

__device__ inline f32x8 wmma_bf16(const Frag& a, const Frag& b, f32x8 c) {
  return __builtin_amdgcn_wmma_f32_16x16x32_bf16(false, a.v, false, b.v,
                                                 (short)0, c, false, false);
}

// ---------------- cos/sin MROPE table: cs[s][j] = (cos, sin) ----------------
__global__ __launch_bounds__(256) void k_cs(const int* __restrict__ pos,
                                            float2* __restrict__ cs) {
  int idx = blockIdx.x * 256 + threadIdx.x;   // [0, 4096*128)
  int s = idx >> 7, j = idx & 127;
  int sect = (j < 32) ? 0 : ((j < 80) ? 1 : 2);   // MROPE sections 32/48/48
  float p = (float)pos[sect * 4096 + s];
  float fi = (float)(j & 63);
  float inv = __expf(fi * (-0.21586735f));        // ln(1e6)/64
  float a = p * inv;
  cs[idx] = make_float2(__cosf(a), __sinf(a));
}

// ---------------- RMSNorm -> bf16, one wave per row of 2048 ----------------
__global__ __launch_bounds__(256) void k_rms(const float* __restrict__ hs,
                                             const float* __restrict__ lnw,
                                             unsigned short* __restrict__ xbf) {
  int row  = blockIdx.x * 8 + (threadIdx.x >> 5);   // [0, L*S)
  int lane = threadIdx.x & 31;
  int l = row >> 12;
  const float* r = hs  + (size_t)row * 2048;
  const float* g = lnw + (size_t)l * 2048;
  float4 v[16];
  float ss = 0.f;
#pragma unroll
  for (int c = 0; c < 16; ++c) {
    v[c] = *(const float4*)(r + (c * 32 + lane) * 4);
    ss += v[c].x * v[c].x + v[c].y * v[c].y + v[c].z * v[c].z + v[c].w * v[c].w;
  }
#pragma unroll
  for (int m = 16; m >= 1; m >>= 1) ss += __shfl_xor(ss, m, 32);
  float rstd = rsqrtf(ss * (1.0f / 2048.0f) + 1e-6f);
  unsigned short* o = xbf + (size_t)row * 2048;
#pragma unroll
  for (int c = 0; c < 16; ++c) {
    float4 gw = *(const float4*)(g + (c * 32 + lane) * 4);
    uint2 st;
    st.x = (unsigned)f2bf(v[c].x * rstd * gw.x) | ((unsigned)f2bf(v[c].y * rstd * gw.y) << 16);
    st.y = (unsigned)f2bf(v[c].z * rstd * gw.z) | ((unsigned)f2bf(v[c].w * rstd * gw.w) << 16);
    *(uint2*)(o + (c * 32 + lane) * 4) = st;
  }
}

// ------- repack fp32 weights (D x N row-major) into WMMA B fragments -------
// One block = one 32x16 fragment. Contiguous k order, matching the A loads:
//   lane n<16 : N=n,      VGPR v holds K = 2v, 2v+1
//   lane n>=16: N=n-16,   VGPR v holds K = 16+2v, 16+2v+1
__global__ __launch_bounds__(256) void k_pack(const float* __restrict__ w,
                                              unsigned* __restrict__ out, int N) {
  int frag = blockIdx.x;                 // ((l*64 + kc)*NT + nt)
  int tid  = threadIdx.x;
  int vv   = tid & 7;
  int lane = tid >> 3;
  int NT = N >> 4;
  int nt = frag % NT;
  int kc = (frag / NT) & 63;
  int l  = frag / (NT * 64);
  int n = nt * 16 + (lane & 15);
  int k = kc * 32 + ((lane >> 4) << 4) + 2 * vv;
  const float* wl = w + (size_t)l * 2048 * N;
  unsigned lo16 = f2bf(wl[(size_t)k * N + n]);
  unsigned hi16 = f2bf(wl[(size_t)(k + 1) * N + n]);
  out[(size_t)frag * 256 + tid] = lo16 | (hi16 << 16);
}

// ---- K projection GEMM (4096x256 @ K=2048 per layer) + bias + RoPE -> Kr ----
// Block: 8 waves, each wave owns 16 rows x all 256 cols (16 C tiles).
__global__ __launch_bounds__(256) void k_kproj(const unsigned short* __restrict__ xbf,
                                               const unsigned* __restrict__ kpack,
                                               const float* __restrict__ kb,
                                               const float2* __restrict__ cs,
                                               unsigned short* __restrict__ Kr) {
  int l = blockIdx.x >> 5;
  int mblk = blockIdx.x & 31;
  int wv = threadIdx.x >> 5, lane = threadIdx.x & 31;
  int lo = lane & 15, hi = lane >> 4;
  int m0 = mblk * 128 + wv * 16;
  const unsigned short* xr = xbf + ((size_t)l * 4096 + m0 + lo) * 2048;
  const unsigned* bp = kpack + (size_t)l * 64 * 16 * 256 + lane * 8;
  f32x8 acc[16];
#pragma unroll
  for (int t = 0; t < 16; ++t)
#pragma unroll
    for (int r = 0; r < 8; ++r) acc[t][r] = 0.f;

  for (int kc = 0; kc < 64; ++kc) {
    Frag a;   // contiguous k order (k = 16*hi + s): two contiguous 16B k-chunks per lane
    a.u[0] = *(const uint4*)(xr + kc * 32 + hi * 16);
    a.u[1] = *(const uint4*)(xr + kc * 32 + hi * 16 + 8);
    const unsigned* bq = bp + kc * (16 * 256);
    __builtin_prefetch(bq + 16 * 256, 0, 3);   // keep-near prefetch of next B panel
#pragma unroll
    for (int nt = 0; nt < 16; ++nt) {
      Frag b;
      b.u[0] = *(const uint4*)(bq + nt * 256);
      b.u[1] = *(const uint4*)(bq + nt * 256 + 4);
      acc[nt] = wmma_bf16(a, b, acc[nt]);
    }
  }
  // bias
#pragma unroll
  for (int nt = 0; nt < 16; ++nt) {
    float bb = kb[l * 256 + nt * 16 + lo];
#pragma unroll
    for (int r = 0; r < 8; ++r) acc[nt][r] += bb;
  }
  // RoPE fully in-register: partner column (+/-64) is tile t+/-4, same lane/vgpr
#pragma unroll
  for (int head = 0; head < 2; ++head) {
#pragma unroll
    for (int i = 0; i < 4; ++i) {
      int t1 = head * 8 + i, t2 = t1 + 4;
      int jj1 = i * 16 + lo, jj2 = jj1 + 64;
#pragma unroll
      for (int r = 0; r < 8; ++r) {
        int s = m0 + r + hi * 8;
        float2 c1 = cs[s * 128 + jj1];
        float2 c2 = cs[s * 128 + jj2];
        float x1 = acc[t1][r], x2 = acc[t2][r];
        size_t base = (((size_t)l * 2 + head) * 4096 + s) * 128;
        Kr[base + jj1] = f2bf(x1 * c1.x - x2 * c1.y);
        Kr[base + jj2] = f2bf(x2 * c2.x + x1 * c2.y);
      }
    }
  }
}

// ---- Q projection, K-split: grid = L x 4(nb: 512 cols) x 4(ks: 16 kc each) ----
// Wave: m-subtile (w&1), one head (w>>1). Accumulates 16 of 64 K-steps into
// fp32 partials pq[ks][l][q][n]; epilogue kernel reduces + bias + RoPE.
__global__ __launch_bounds__(256) void k_qproj(const unsigned short* __restrict__ xbf,
                                               const unsigned* __restrict__ qpack,
                                               const int* __restrict__ qidx,
                                               float* __restrict__ pq) {
  int ks = blockIdx.x & 3;
  int nb = (blockIdx.x >> 2) & 3;
  int l  = blockIdx.x >> 4;
  int wv = threadIdx.x >> 5, lane = threadIdx.x & 31;
  int lo = lane & 15, hi = lane >> 4;
  int msub = wv & 1;
  int head = nb * 4 + (wv >> 1);
  int srow_a = qidx[msub * 16 + lo];     // gathered query row
  const unsigned short* xr = xbf + ((size_t)l * 4096 + srow_a) * 2048;
  const unsigned* bp = qpack + (size_t)l * 64 * 128 * 256 + lane * 8;
  f32x8 acc[8];
#pragma unroll
  for (int t = 0; t < 8; ++t)
#pragma unroll
    for (int r = 0; r < 8; ++r) acc[t][r] = 0.f;

  for (int kc = ks * 16; kc < ks * 16 + 16; ++kc) {
    Frag a;
    a.u[0] = *(const uint4*)(xr + kc * 32 + hi * 16);
    a.u[1] = *(const uint4*)(xr + kc * 32 + hi * 16 + 8);
    const unsigned* bq = bp + (size_t)kc * 128 * 256 + (head * 8) * 256;
#pragma unroll
    for (int nt = 0; nt < 8; ++nt) {
      Frag b;
      b.u[0] = *(const uint4*)(bq + nt * 256);
      b.u[1] = *(const uint4*)(bq + nt * 256 + 4);
      acc[nt] = wmma_bf16(a, b, acc[nt]);
    }
  }
  // store fp32 partials
#pragma unroll
  for (int nt = 0; nt < 8; ++nt) {
#pragma unroll
    for (int r = 0; r < 8; ++r) {
      int q = msub * 16 + r + hi * 8;
      int n = head * 128 + nt * 16 + lo;
      pq[(((size_t)ks * 8 + l) * 32 + q) * 2048 + n] = acc[nt][r];
    }
  }
}

// ---- Q epilogue: reduce 4 K-split partials + bias + RoPE -> bf16 Qr ----
__global__ __launch_bounds__(256) void k_qrope(const float* __restrict__ pq,
                                               const float* __restrict__ qb,
                                               const int* __restrict__ qidx,
                                               const float2* __restrict__ cs,
                                               unsigned short* __restrict__ Qr) {
  int idx = blockIdx.x * 256 + threadIdx.x;   // [0, 8*32*16*64)
  int j0   = idx & 63;
  int head = (idx >> 6) & 15;
  int q    = (idx >> 10) & 31;
  int l    = idx >> 15;
  int n1 = head * 128 + j0;
  int n2 = n1 + 64;
  size_t base = ((size_t)l * 32 + q) * 2048;
  const size_t kstride = (size_t)8 * 32 * 2048;
  float v1 = 0.f, v2 = 0.f;
#pragma unroll
  for (int ks = 0; ks < 4; ++ks) {
    v1 += pq[ks * kstride + base + n1];
    v2 += pq[ks * kstride + base + n2];
  }
  v1 += qb[l * 2048 + n1];
  v2 += qb[l * 2048 + n2];
  int s = qidx[q];
  float2 c1 = cs[s * 128 + j0];
  float2 c2 = cs[s * 128 + j0 + 64];
  size_t ob = (((size_t)l * 16 + head) * 32 + q) * 128;
  Qr[ob + j0]      = f2bf(v1 * c1.x - v2 * c1.y);
  Qr[ob + j0 + 64] = f2bf(v2 * c2.x + v1 * c2.y);
}

// -------- attention: scores = Q @ K^T, scale, causal mask, softmax(S) --------
// Grid: L*H*2 blocks x 512 threads; block: 16 q-rows x full S=4096.
// 16 waves x 256 cols each; scores stay register-resident through the whole
// softmax (mask -> max -> exp -> sum -> normalize); global written exactly once.
__global__ __launch_bounds__(512) void k_attn(const unsigned short* __restrict__ Qr,
                                              const unsigned short* __restrict__ Kr,
                                              const int* __restrict__ qidx,
                                              float* __restrict__ out) {
  __shared__ float ldsP[16][16];
  __shared__ float redM[16];
  __shared__ float redS[16];
  int bx = blockIdx.x;
  int qh = bx & 1;
  int h  = (bx >> 1) & 15;
  int l  = bx >> 5;
  int kv = h >> 3;
  int wv = threadIdx.x >> 5, lane = threadIdx.x & 31;
  int lo = lane & 15, hi = lane >> 4;
  int qbase = qh * 16;

  // preload all 4 A (query) fragments — reused for every column tile
  const unsigned short* qr = Qr + (((size_t)l * 16 + h) * 32 + qbase + lo) * 128;
  Frag aq[4];
#pragma unroll
  for (int kc = 0; kc < 4; ++kc) {
    aq[kc].u[0] = *(const uint4*)(qr + kc * 32 + hi * 16);
    aq[kc].u[1] = *(const uint4*)(qr + kc * 32 + hi * 16 + 8);
  }
  int qidx_r[8];
#pragma unroll
  for (int r = 0; r < 8; ++r) qidx_r[r] = qidx[qbase + r + hi * 8];

  const unsigned short* kr = Kr + ((size_t)l * 2 + kv) * 4096 * 128;
  float* outp = out + (((size_t)l * 16 + h) * 32 + qbase) * 4096;
  const float scale = 0.08838834764831845f;   // 1/sqrt(128)

  f32x8 acc[16];
  float rmax[8];
#pragma unroll
  for (int r = 0; r < 8; ++r) rmax[r] = -3.4e38f;

  // scores for this wave's 256 columns, kept in registers
#pragma unroll
  for (int ntl = 0; ntl < 16; ++ntl) {
    int s0 = wv * 256 + ntl * 16;
    const unsigned short* krow = kr + (size_t)(s0 + lo) * 128;
    f32x8 a;
#pragma unroll
    for (int r = 0; r < 8; ++r) a[r] = 0.f;
#pragma unroll
    for (int kc = 0; kc < 4; ++kc) {
      Frag b;      // B fragment: lane's 16 contiguous hd values of row s0+lo
      int kb0 = kc * 32 + hi * 16;
      b.u[0] = *(const uint4*)(krow + kb0);
      b.u[1] = *(const uint4*)(krow + kb0 + 8);
      a = wmma_bf16(aq[kc], b, a);
    }
    int s = s0 + lo;
#pragma unroll
    for (int r = 0; r < 8; ++r) {
      float v = a[r] * scale + ((s > qidx_r[r]) ? -1.0e9f : 0.f);
      a[r] = v;
      rmax[r] = fmaxf(rmax[r], v);
    }
    acc[ntl] = a;
  }
  // row max: shuffle within 16-lane halves, then across the 16 waves via LDS
#pragma unroll
  for (int r = 0; r < 8; ++r) {
#pragma unroll
    for (int m = 8; m >= 1; m >>= 1) rmax[r] = fmaxf(rmax[r], __shfl_xor(rmax[r], m, 32));
  }
  if (lo == 0) {
#pragma unroll
    for (int r = 0; r < 8; ++r) ldsP[wv][hi * 8 + r] = rmax[r];
  }
  __syncthreads();
  if (threadIdx.x < 16) {
    float mx = ldsP[0][threadIdx.x];
#pragma unroll
    for (int ww = 1; ww < 16; ++ww) mx = fmaxf(mx, ldsP[ww][threadIdx.x]);
    redM[threadIdx.x] = mx;
  }
  __syncthreads();

  // exp in registers + row sums
  float mrow[8], psum[8];
#pragma unroll
  for (int r = 0; r < 8; ++r) { mrow[r] = redM[r + hi * 8]; psum[r] = 0.f; }
#pragma unroll
  for (int ntl = 0; ntl < 16; ++ntl) {
#pragma unroll
    for (int r = 0; r < 8; ++r) {
      float e = __expf(acc[ntl][r] - mrow[r]);
      acc[ntl][r] = e;
      psum[r] += e;
    }
  }
#pragma unroll
  for (int r = 0; r < 8; ++r) {
#pragma unroll
    for (int m = 8; m >= 1; m >>= 1) psum[r] += __shfl_xor(psum[r], m, 32);
  }
  if (lo == 0) {
#pragma unroll
    for (int r = 0; r < 8; ++r) ldsP[wv][hi * 8 + r] = psum[r];
  }
  __syncthreads();
  if (threadIdx.x < 16) {
    float sm = 0.f;
#pragma unroll
    for (int ww = 0; ww < 16; ++ww) sm += ldsP[ww][threadIdx.x];
    redS[threadIdx.x] = 1.0f / sm;
  }
  __syncthreads();

  // normalize and single write to global
  float inv[8];
#pragma unroll
  for (int r = 0; r < 8; ++r) inv[r] = redS[r + hi * 8];
#pragma unroll
  for (int ntl = 0; ntl < 16; ++ntl) {
    int s = wv * 256 + ntl * 16 + lo;
#pragma unroll
    for (int r = 0; r < 8; ++r) {
      outp[(size_t)(r + hi * 8) * 4096 + s] = acc[ntl][r] * inv[r];
    }
  }
}

extern "C" void kernel_launch(void* const* d_in, const int* in_sizes, int n_in,
                              void* d_out, int out_size, void* d_ws, size_t ws_size,
                              hipStream_t stream) {
  (void)in_sizes; (void)n_in; (void)out_size; (void)ws_size;
  const float* hs  = (const float*)d_in[0];
  const float* lnw = (const float*)d_in[1];
  const float* qw  = (const float*)d_in[2];
  const float* qb  = (const float*)d_in[3];
  const float* kw  = (const float*)d_in[4];
  const float* kb  = (const float*)d_in[5];
  const int*   pos = (const int*)d_in[6];
  const int*   qix = (const int*)d_in[7];
  float* out = (float*)d_out;

  char* w = (char*)d_ws;
  float2* cs = (float2*)w;                   w += (size_t)4096 * 128 * sizeof(float2);
  unsigned short* xbf = (unsigned short*)w;  w += (size_t)8 * 4096 * 2048 * 2;
  unsigned* kpack = (unsigned*)w;            w += (size_t)8 * 64 * 16 * 256 * 4;
  unsigned* qpack = (unsigned*)w;            w += (size_t)8 * 64 * 128 * 256 * 4;
  unsigned short* Kr = (unsigned short*)w;   w += (size_t)8 * 2 * 4096 * 128 * 2;
  unsigned short* Qr = (unsigned short*)w;   w += (size_t)8 * 16 * 32 * 128 * 2;
  float* pq = (float*)w;

  k_cs   <<<2048,  256, 0, stream>>>(pos, cs);
  k_rms  <<<4096,  256, 0, stream>>>(hs, lnw, xbf);
  k_pack <<<8192,  256, 0, stream>>>(kw, kpack, 256);
  k_pack <<<65536, 256, 0, stream>>>(qw, qpack, 2048);
  k_kproj<<<256,   256, 0, stream>>>(xbf, kpack, kb, cs, Kr);
  k_qproj<<<128,   256, 0, stream>>>(xbf, qpack, qix, pq);
  k_qrope<<<1024,  256, 0, stream>>>(pq, qb, qix, cs, Qr);
  k_attn <<<256,   512, 0, stream>>>(Qr, Kr, qix, out);
}
